// TransitionUpBlock_85461259256093
// MI455X (gfx1250) — compile-verified, matrix operand order checked by reading, not run on hardware
//
#include <hip/hip_runtime.h>
#include <math.h>

// MI455X (gfx1250) implementation of TransitionUp + PointTransformerBlock.
// All dense 128-wide GEMMs run on V_WMMA_F32_16X16X32_BF16 with fused
// input-BN+activation and fused output-BN-stats accumulation. Operands are
// staged in LDS in *fragment order* (one aligned 32B record per lane per
// fragment -> 2x ds_load_b128 straight into WMMA source registers), and the
// operand feed is software-pipelined: all 8 B-fragments of a k-tile plus the
// next k-tile's A-fragment are issued before the 8-WMMA chain consumes them,
// so LDS latency overlaps matrix-core execution (partial s_wait_dscnt).

typedef __attribute__((ext_vector_type(16))) __bf16 v16bf;
typedef __attribute__((ext_vector_type(2)))  __bf16 v2bf;
typedef __attribute__((ext_vector_type(8)))  float  v8f;

#define BN_EPS 1e-5f

__device__ __forceinline__ float gelu_f(float x) {
  return 0.5f * x * (1.0f + erff(x * 0.70710678118654752440f));
}

// Fragment-order dword index helpers. ISA 16-bit operand layout: dword v of
// lane L holds k = (v>>2)*16 + (L>>4)*8 + (v&3)*2 (+0/+1 packed).
__device__ __forceinline__ int frag_dword(int kk /*k&31*/) {
  return ((kk >> 4) << 2) + ((kk >> 1) & 3);
}
__device__ __forceinline__ int frag_lane_half(int kk) { return (kk >> 3) & 1; }

// --------------------------------------------------------------------------
// Y[M,128] = act(bn(X))[M,K] @ W[K,128] + bias   (K in {128,256}, M % 64 == 0)
// in_mode: 0 raw, 1 bn+relu, 2 bn+gelu (bn derived from in_stats sum/sumsq)
// out_stats: optional per-channel {sum[0:128], sumsq[128:256]} accumulation.
// Block = 128 threads = 4 waves; each wave owns a 16-row strip, 8 n-tiles.
// --------------------------------------------------------------------------
__global__ void __launch_bounds__(128)
gemm_bn_wmma(const float* __restrict__ X, const float* __restrict__ W,
             const float* __restrict__ bias,
             const float* __restrict__ in_stats, const float* __restrict__ in_g,
             const float* __restrict__ in_b, float inv_cnt_in, int in_mode,
             float* __restrict__ Y, float* __restrict__ out_stats,
             int M, int K)
{
  // W fragments: [kt4][nt][lane][8 dwords]  -> 4*8*32*8 dwords = 32 KB
  __shared__ __attribute__((aligned(32))) unsigned int sWd[4 * 8 * 32 * 8];
  // A fragments: [wave][kt4][lane][8 dwords] -> 4*4*32*8 dwords = 16 KB
  __shared__ __attribute__((aligned(32))) unsigned int sAd[4 * 4 * 32 * 8];
  __shared__ float sScale[128];
  __shared__ float sBias[128];
  __shared__ float sSum[128];
  __shared__ float sSq[128];

  const int tid  = threadIdx.x;
  const int wave = tid >> 5;
  const int lane = tid & 31;
  const int half = lane >> 4;
  const int m0   = blockIdx.x * 64 + wave * 16;

  if (tid < 128) { sSum[tid] = 0.0f; sSq[tid] = 0.0f; }

  v8f acc[8] = {};

  for (int kc = 0; kc < K; kc += 128) {
    __syncthreads();   // protect LDS reuse from previous chunk (+init)
    // per-channel input BN transform for this K-chunk
    {
      float sc = 1.0f, bs = 0.0f;
      if (in_mode != 0) {
        int c = kc + tid;
        float mean = in_stats[c] * inv_cnt_in;
        float var  = in_stats[128 + c] * inv_cnt_in - mean * mean;
        float rs   = rsqrtf(var + BN_EPS);
        sc = rs * in_g[c];
        bs = in_b[c] - mean * sc;
      }
      sScale[tid] = sc; sBias[tid] = bs;
    }
    __syncthreads();

    // Stage W chunk into fragment order (bf16 pairs packed into dwords).
    for (int i = tid; i < 8192; i += 128) {
      int k2 = i >> 7;            // k = 2*k2, 0..126
      int n  = i & 127;
      int k  = k2 << 1;
      float w0 = W[((size_t)(kc + k)) * 128 + n];
      float w1 = W[((size_t)(kc + k + 1)) * 128 + n];
      int kk = k & 31;
      int off = ((((k >> 5) * 8 + (n >> 4)) * 32) +
                 ((n & 15) + (frag_lane_half(kk) << 4))) * 8 + frag_dword(kk);
      v2bf p; p[0] = (__bf16)w0; p[1] = (__bf16)w1;
      ((v2bf*)sWd)[off] = p;
    }
    // Stage X tile (64 rows x 128 cols) coalesced, transform, fragment order.
    for (int i = tid; i < 4096; i += 128) {
      int r  = i >> 6;            // 0..63 local row
      int k  = (i & 63) << 1;     // 0..126 even
      const float* xp = X + (size_t)(blockIdx.x * 64 + r) * K + kc + k;
      float x0 = xp[0] * sScale[k]     + sBias[k];
      float x1 = xp[1] * sScale[k + 1] + sBias[k + 1];
      if (in_mode == 1)      { x0 = fmaxf(x0, 0.0f); x1 = fmaxf(x1, 0.0f); }
      else if (in_mode == 2) { x0 = gelu_f(x0);      x1 = gelu_f(x1); }
      int kk = k & 31;
      int off = ((((r >> 4) * 4 + (k >> 5)) * 32) +
                 ((r & 15) + (frag_lane_half(kk) << 4))) * 8 + frag_dword(kk);
      v2bf p; p[0] = (__bf16)x0; p[1] = (__bf16)x1;
      ((v2bf*)sAd)[off] = p;
    }
    __syncthreads();

    // Software-pipelined WMMA chain: all B-fragments of a k-tile plus the
    // next k-tile's A-fragment are in flight before the chain consumes them.
    v16bf a_cur = *reinterpret_cast<const v16bf*>(
        &sAd[(((wave * 4 + 0) * 32) + lane) * 8]);
#pragma unroll
    for (int kt4 = 0; kt4 < 4; ++kt4) {
      v16bf bf[8];
#pragma unroll
      for (int nt = 0; nt < 8; ++nt)
        bf[nt] = *reinterpret_cast<const v16bf*>(
            &sWd[(((kt4 * 8 + nt) * 32) + lane) * 8]);
      v16bf a_next = a_cur;
      if (kt4 < 3)
        a_next = *reinterpret_cast<const v16bf*>(
            &sAd[(((wave * 4 + kt4 + 1) * 32) + lane) * 8]);
#pragma unroll
      for (int nt = 0; nt < 8; ++nt)
        acc[nt] = __builtin_amdgcn_wmma_f32_16x16x32_bf16(
            false, a_cur, false, bf[nt], (short)0, acc[nt], false, false);
      a_cur = a_next;
    }
  }

  // Epilogue: bias add, store, per-channel stats (lane holds one column,
  // 8 rows: m = m0 + half*8 + j per the f32 C/D layout).
#pragma unroll
  for (int nt = 0; nt < 8; ++nt) {
    int n = (nt << 4) + (lane & 15);
    float bv = bias[n];
    float s = 0.0f, s2 = 0.0f;
#pragma unroll
    for (int j = 0; j < 8; ++j) {
      int m = m0 + (half << 3) + j;
      float v = acc[nt][j] + bv;
      Y[(size_t)m * 128 + n] = v;
      s += v; s2 += v * v;
    }
    if (out_stats) { atomicAdd(&sSum[n], s); atomicAdd(&sSq[n], s2); }
  }
  if (out_stats) {
    __syncthreads();
    if (tid < 128) {
      atomicAdd(&out_stats[tid],       sSum[tid]);
      atomicAdd(&out_stats[128 + tid], sSq[tid]);
    }
  }
}

// ---- 3-NN of points1 (8192) against points2 (2048), trilinear weights -----
__global__ void __launch_bounds__(256)
knn3_kernel(const float* __restrict__ qpts, const float* __restrict__ rpts,
            int* __restrict__ idx3, float* __restrict__ w3)
{
  __shared__ float rx[2048], ry[2048], rz[2048];
  const int b = blockIdx.y;
  const float* R = rpts + (size_t)b * 2048 * 3;
  for (int i = threadIdx.x; i < 2048; i += 256) {
    rx[i] = R[i * 3 + 0]; ry[i] = R[i * 3 + 1]; rz[i] = R[i * 3 + 2];
  }
  __syncthreads();
  const int qi = blockIdx.x * 256 + threadIdx.x;
  const float* Q = qpts + ((size_t)b * 8192 + qi) * 3;
  float qx = Q[0], qy = Q[1], qz = Q[2];
  float bd[3] = {3.4e38f, 3.4e38f, 3.4e38f};
  int   bi[3] = {0, 0, 0};
  for (int i = 0; i < 2048; ++i) {
    float dx = qx - rx[i], dy = qy - ry[i], dz = qz - rz[i];
    float d = dx * dx + dy * dy + dz * dz;
    if (d < bd[2]) {
      float cd = d; int ci = i;
#pragma unroll
      for (int j = 0; j < 3; ++j)
        if (cd < bd[j]) { float td = bd[j]; int ti = bi[j];
                          bd[j] = cd; bi[j] = ci; cd = td; ci = ti; }
    }
  }
  float w0 = 1.0f / (sqrtf(bd[0]) + 1e-8f);
  float w1 = 1.0f / (sqrtf(bd[1]) + 1e-8f);
  float w2 = 1.0f / (sqrtf(bd[2]) + 1e-8f);
  float wsum = w0 + w1 + w2;
  size_t o = ((size_t)b * 8192 + qi) * 3;
  idx3[o] = bi[0]; idx3[o + 1] = bi[1]; idx3[o + 2] = bi[2];
  w3[o] = w0 / wsum; w3[o + 1] = w1 / wsum; w3[o + 2] = w2 / wsum;
}

// ---- 16-NN self-KNN over points1 (8192 per batch) -------------------------
__global__ void __launch_bounds__(256)
knn16_kernel(const float* __restrict__ pts, int* __restrict__ idx16)
{
  __shared__ float rx[2048], ry[2048], rz[2048];
  const int b = blockIdx.y;
  const int qi = blockIdx.x * 256 + threadIdx.x;
  const float* Q = pts + ((size_t)b * 8192 + qi) * 3;
  float qx = Q[0], qy = Q[1], qz = Q[2];
  float bd[16]; int bi[16];
#pragma unroll
  for (int j = 0; j < 16; ++j) { bd[j] = 3.4e38f; bi[j] = 0; }
  for (int t = 0; t < 8192; t += 2048) {
    __syncthreads();
    const float* R = pts + ((size_t)b * 8192 + t) * 3;
    for (int i = threadIdx.x; i < 2048; i += 256) {
      rx[i] = R[i * 3]; ry[i] = R[i * 3 + 1]; rz[i] = R[i * 3 + 2];
    }
    __syncthreads();
    for (int i = 0; i < 2048; ++i) {
      float dx = qx - rx[i], dy = qy - ry[i], dz = qz - rz[i];
      float d = dx * dx + dy * dy + dz * dz;
      if (d < bd[15]) {
        float cd = d; int ci = t + i;
#pragma unroll
        for (int j = 0; j < 16; ++j)
          if (cd < bd[j]) { float td = bd[j]; int ti = bi[j];
                            bd[j] = cd; bi[j] = ci; cd = td; ci = ti; }
      }
    }
  }
  size_t o = ((size_t)b * 8192 + qi) * 16;
#pragma unroll
  for (int j = 0; j < 16; ++j) idx16[o + j] = bi[j];
}

// ---- feats = relu(bn(t1)) + sum_j w3_j * relu(bn(t2))[idx3_j] -------------
__global__ void __launch_bounds__(128)
combine_up(const float* __restrict__ t1, const float* __restrict__ st1,
           const float* __restrict__ g1, const float* __restrict__ b1, float invc1,
           const float* __restrict__ t2, const float* __restrict__ st2,
           const float* __restrict__ g2, const float* __restrict__ b2, float invc2,
           const int* __restrict__ idx3, const float* __restrict__ w3,
           float* __restrict__ feats)
{
  int c = threadIdx.x;
  int bn = blockIdx.x;
  int b = bn >> 13;
  float m1 = st1[c] * invc1, v1 = st1[128 + c] * invc1 - m1 * m1;
  float r1 = rsqrtf(v1 + BN_EPS);
  float sc1 = r1 * g1[c], bs1 = b1[c] - m1 * sc1;
  float m2 = st2[c] * invc2, v2 = st2[128 + c] * invc2 - m2 * m2;
  float r2 = rsqrtf(v2 + BN_EPS);
  float sc2 = r2 * g2[c], bs2 = b2[c] - m2 * sc2;
  float f = fmaxf(t1[(size_t)bn * 128 + c] * sc1 + bs1, 0.0f);
  size_t o3 = (size_t)bn * 3;
#pragma unroll
  for (int j = 0; j < 3; ++j) {
    int id = idx3[o3 + j];
    float w = w3[o3 + j];
    float f2 = fmaxf(t2[((size_t)(b * 2048 + id)) * 128 + c] * sc2 + bs2, 0.0f);
    f += w * f2;
  }
  feats[(size_t)bn * 128 + c] = f;
}

// ---- pos1 = (p - p[knn]) @ Wd1 + bd1, with bn_delta stats -----------------
__global__ void __launch_bounds__(256)
pos1_kernel(const float* __restrict__ pts, const int* __restrict__ idx16,
            const float* __restrict__ Wd1, const float* __restrict__ bd1,
            float* __restrict__ pos1, float* __restrict__ st)
{
  __shared__ float s[6];
  if (threadIdx.x < 6) s[threadIdx.x] = 0.0f;
  __syncthreads();
  int r = blockIdx.x * 256 + threadIdx.x;    // (b,n,k) row, < 262144
  int bn = r >> 4;
  int b = bn >> 13;
  int j = idx16[r];
  const float* P  = pts + (size_t)bn * 3;
  const float* Pj = pts + ((size_t)(b * 8192 + j)) * 3;
  float dx = P[0] - Pj[0], dy = P[1] - Pj[1], dz = P[2] - Pj[2];
  float o0 = dx * Wd1[0] + dy * Wd1[3] + dz * Wd1[6] + bd1[0];
  float o1 = dx * Wd1[1] + dy * Wd1[4] + dz * Wd1[7] + bd1[1];
  float o2 = dx * Wd1[2] + dy * Wd1[5] + dz * Wd1[8] + bd1[2];
  size_t o = (size_t)r * 3;
  pos1[o] = o0; pos1[o + 1] = o1; pos1[o + 2] = o2;
  atomicAdd(&s[0], o0); atomicAdd(&s[1], o1); atomicAdd(&s[2], o2);
  atomicAdd(&s[3], o0 * o0); atomicAdd(&s[4], o1 * o1); atomicAdd(&s[5], o2 * o2);
  __syncthreads();
  if (threadIdx.x < 3) {
    atomicAdd(&st[threadIdx.x],       s[threadIdx.x]);
    atomicAdd(&st[128 + threadIdx.x], s[threadIdx.x + 3]);
  }
}

// ---- in-place pos1 <- gelu(bn_delta(pos1)) --------------------------------
__global__ void __launch_bounds__(256)
pos1g_kernel(float* __restrict__ p, const float* __restrict__ st,
             const float* __restrict__ g, const float* __restrict__ bb, float invc)
{
  int i = blockIdx.x * 256 + threadIdx.x;    // < 786432
  int c = i % 3;
  float mean = st[c] * invc, var = st[128 + c] * invc - mean * mean;
  float rs = rsqrtf(var + BN_EPS);
  p[i] = gelu_f((p[i] - mean) * rs * g[c] + bb[c]);
}

// ---- gamma = q - key[idx] + pos   (pos recomputed: pos1g @ Wd2 + bd2) -----
__global__ void __launch_bounds__(128)
gamma_assemble(const float* __restrict__ qb, const float* __restrict__ keyb,
               const int* __restrict__ idx16, const float* __restrict__ p1g,
               const float* __restrict__ Wd2, const float* __restrict__ bd2,
               float* __restrict__ out, float* __restrict__ st)
{
  int c = threadIdx.x;
  float w0 = Wd2[c], w1 = Wd2[128 + c], w2 = Wd2[256 + c], bl = bd2[c];
  float s = 0.0f, s2 = 0.0f;
  for (int u = 0; u < 8; ++u) {
    int bn = blockIdx.x * 8 + u;
    int b = bn >> 13;
    float qv = qb[(size_t)bn * 128 + c];
    for (int k = 0; k < 16; ++k) {
      int r = bn * 16 + k;
      int j = idx16[r];
      float kv = keyb[((size_t)(b * 8192 + j)) * 128 + c];
      const float* pg = p1g + (size_t)r * 3;
      float pos = pg[0] * w0 + pg[1] * w1 + pg[2] * w2 + bl;
      float g = qv - kv + pos;
      out[(size_t)r * 128 + c] = g;
      s += g; s2 += g * g;
    }
  }
  atomicAdd(&st[c], s);
  atomicAdd(&st[128 + c], s2);
}

// ---- softmax over k (per channel) + einsum with value[idx] + pos ----------
__global__ void __launch_bounds__(128)
attn_einsum(const float* __restrict__ gamma, const float* __restrict__ valb,
            const int* __restrict__ idx16, const float* __restrict__ p1g,
            const float* __restrict__ Wd2, const float* __restrict__ bd2,
            float* __restrict__ out, float* __restrict__ st)
{
  int c = threadIdx.x;
  float w0 = Wd2[c], w1 = Wd2[128 + c], w2 = Wd2[256 + c], bl = bd2[c];
  float s = 0.0f, s2 = 0.0f;
  for (int u = 0; u < 8; ++u) {
    int bn = blockIdx.x * 8 + u;
    int b = bn >> 13;
    float g[16];
    float mx = -3.4e38f;
#pragma unroll
    for (int k = 0; k < 16; ++k) {
      g[k] = gamma[((size_t)bn * 16 + k) * 128 + c];
      mx = fmaxf(mx, g[k]);
    }
    float se = 0.0f;
#pragma unroll
    for (int k = 0; k < 16; ++k) { g[k] = expf(g[k] - mx); se += g[k]; }
    float inv = 1.0f / se;
    float o = 0.0f;
#pragma unroll
    for (int k = 0; k < 16; ++k) {
      int r = bn * 16 + k;
      int j = idx16[r];
      const float* pg = p1g + (size_t)r * 3;
      float pos = pg[0] * w0 + pg[1] * w1 + pg[2] * w2 + bl;
      float val = valb[((size_t)(b * 8192 + j)) * 128 + c] + pos;
      o += g[k] * inv * val;
    }
    out[(size_t)bn * 128 + c] = o;
    s += o; s2 += o * o;
  }
  atomicAdd(&st[c], s);
  atomicAdd(&st[128 + c], s2);
}

// ---- d_out = [ feats + gelu(bn3(fc2o)) , points1 ] ------------------------
__global__ void __launch_bounds__(256)
final_kernel(const float* __restrict__ feats, const float* __restrict__ fc2o,
             const float* __restrict__ st, const float* __restrict__ g,
             const float* __restrict__ bb, float invc,
             const float* __restrict__ pts, float* __restrict__ out)
{
  int i = blockIdx.x * 256 + threadIdx.x;
  const int NF = 2 * 8192 * 128;
  if (i < NF) {
    int c = i & 127;
    float mean = st[c] * invc, var = st[128 + c] * invc - mean * mean;
    float rs = rsqrtf(var + BN_EPS);
    out[i] = feats[i] + gelu_f((fc2o[i] - mean) * rs * g[c] + bb[c]);
  } else if (i < NF + 2 * 8192 * 3) {
    out[i] = pts[i - NF];
  }
}

extern "C" void kernel_launch(void* const* d_in, const int* in_sizes, int n_in,
                              void* d_out, int out_size, void* d_ws, size_t ws_size,
                              hipStream_t stream)
{
  (void)in_sizes; (void)n_in; (void)out_size; (void)ws_size;

  const float* feats1  = (const float*)d_in[0];
  const float* points1 = (const float*)d_in[1];
  const float* feats2  = (const float*)d_in[2];
  const float* points2 = (const float*)d_in[3];
  const float *Wf1 = (const float*)d_in[4],  *bf1 = (const float*)d_in[5];
  const float *gf1 = (const float*)d_in[6],  *nf1 = (const float*)d_in[7];
  const float *Wf2 = (const float*)d_in[8],  *bf2 = (const float*)d_in[9];
  const float *gf2 = (const float*)d_in[10], *nf2 = (const float*)d_in[11];
  const float *W1  = (const float*)d_in[12], *bb1 = (const float*)d_in[13];
  const float *g1  = (const float*)d_in[14], *n1  = (const float*)d_in[15];
  const float *W2  = (const float*)d_in[16], *bb2 = (const float*)d_in[17];
  const float *g2  = (const float*)d_in[18], *n2  = (const float*)d_in[19];
  const float *g3  = (const float*)d_in[20], *n3  = (const float*)d_in[21];
  const float *Wq  = (const float*)d_in[22], *bq  = (const float*)d_in[23];
  const float *Wk  = (const float*)d_in[24], *bk  = (const float*)d_in[25];
  const float *Wv  = (const float*)d_in[26], *bv  = (const float*)d_in[27];
  const float *Wd1 = (const float*)d_in[28], *bd1 = (const float*)d_in[29];
  const float *gd  = (const float*)d_in[30], *nd  = (const float*)d_in[31];
  const float *Wd2 = (const float*)d_in[32], *bd2 = (const float*)d_in[33];
  const float *Wg1 = (const float*)d_in[34], *bg1 = (const float*)d_in[35];
  const float *gg1 = (const float*)d_in[36], *ng1 = (const float*)d_in[37];
  const float *Wg2 = (const float*)d_in[38], *bg2 = (const float*)d_in[39];
  const float *gg2 = (const float*)d_in[40], *ng2 = (const float*)d_in[41];

  const int M1 = 16384;      // B*N1
  const int M2 = 4096;       // B*N2
  const int MG = 262144;     // B*N1*k

  float* ws    = (float*)d_ws;
  float* stats = ws;                         // 8 slots x 256 floats
  float* st_f1 = stats + 0 * 256;
  float* st_f2 = stats + 1 * 256;
  float* st_b1 = stats + 2 * 256;
  float* st_dl = stats + 3 * 256;
  float* st_g1 = stats + 4 * 256;
  float* st_g2 = stats + 5 * 256;
  float* st_b2 = stats + 6 * 256;
  float* st_b3 = stats + 7 * 256;
  float* t1    = stats + 2048;
  float* t2    = t1    + (size_t)M1 * 128;
  float* feats = t2    + (size_t)M2 * 128;
  float* fc1o  = feats + (size_t)M1 * 128;
  float* qb    = fc1o  + (size_t)M1 * 128;
  float* keyb  = qb    + (size_t)M1 * 128;
  float* valb  = keyb  + (size_t)M1 * 128;
  float* eo    = valb  + (size_t)M1 * 128;
  float* fc2o  = eo    + (size_t)M1 * 128;
  float* p1    = fc2o  + (size_t)M1 * 128;   // MG*3
  float* w3    = p1    + (size_t)MG * 3;     // M1*3
  int*   idx3  = (int*)(w3 + (size_t)M1 * 3);
  int*   idx16 = idx3 + (size_t)M1 * 3;      // MG
  float* gA    = (float*)(idx16 + (size_t)MG);
  float* gB    = gA + (size_t)MG * 128;

  hipMemsetAsync(stats, 0, 2048 * sizeof(float), stream);

  // --- TransitionUp ---
  gemm_bn_wmma<<<M1 / 64, 128, 0, stream>>>(feats1, Wf1, bf1,
      nullptr, nullptr, nullptr, 0.f, 0, t1, st_f1, M1, 128);
  gemm_bn_wmma<<<M2 / 64, 128, 0, stream>>>(feats2, Wf2, bf2,
      nullptr, nullptr, nullptr, 0.f, 0, t2, st_f2, M2, 256);
  knn3_kernel<<<dim3(32, 2), 256, 0, stream>>>(points1, points2, idx3, w3);
  combine_up<<<M1, 128, 0, stream>>>(t1, st_f1, gf1, nf1, 1.0f / M1,
                                     t2, st_f2, gf2, nf2, 1.0f / M2,
                                     idx3, w3, feats);

  // --- PointTransformerBlock ---
  gemm_bn_wmma<<<M1 / 64, 128, 0, stream>>>(feats, W1, bb1,
      nullptr, nullptr, nullptr, 0.f, 0, fc1o, st_b1, M1, 128);
  knn16_kernel<<<dim3(32, 2), 256, 0, stream>>>(points1, idx16);

  // q/key/value on pre-gather activations (linear commutes with gather)
  gemm_bn_wmma<<<M1 / 64, 128, 0, stream>>>(fc1o, Wq, bq,
      st_b1, g1, n1, 1.0f / M1, 2, qb,   nullptr, M1, 128);
  gemm_bn_wmma<<<M1 / 64, 128, 0, stream>>>(fc1o, Wk, bk,
      st_b1, g1, n1, 1.0f / M1, 2, keyb, nullptr, M1, 128);
  gemm_bn_wmma<<<M1 / 64, 128, 0, stream>>>(fc1o, Wv, bv,
      st_b1, g1, n1, 1.0f / M1, 2, valb, nullptr, M1, 128);

  // position encoding (3-channel part, gelu(bn) applied in place)
  pos1_kernel<<<MG / 256, 256, 0, stream>>>(points1, idx16, Wd1, bd1, p1, st_dl);
  pos1g_kernel<<<(MG * 3) / 256, 256, 0, stream>>>(p1, st_dl, gd, nd, 1.0f / MG);

  // gamma MLP (the two big M=262144 WMMA GEMMs, BN+gelu fused into A-loads)
  gamma_assemble<<<M1 / 8, 128, 0, stream>>>(qb, keyb, idx16, p1, Wd2, bd2, gA, st_g1);
  gemm_bn_wmma<<<MG / 64, 128, 0, stream>>>(gA, Wg1, bg1,
      st_g1, gg1, ng1, 1.0f / MG, 2, gB, st_g2, MG, 128);
  gemm_bn_wmma<<<MG / 64, 128, 0, stream>>>(gB, Wg2, bg2,
      st_g2, gg2, ng2, 1.0f / MG, 2, gA, nullptr, MG, 128);

  // softmax over k + weighted sum of (value[idx] + pos)
  attn_einsum<<<M1 / 8, 128, 0, stream>>>(gA, valb, idx16, p1, Wd2, bd2, eo, st_b2);

  // fc2 with bn2+gelu fused in, bn3 stats fused out
  gemm_bn_wmma<<<M1 / 64, 128, 0, stream>>>(eo, W2, bb2,
      st_b2, g2, n2, 1.0f / M1, 2, fc2o, st_b3, M1, 128);

  // residual + bn3 + gelu, plus points passthrough
  int total = M1 * 128 + M1 * 3;
  final_kernel<<<(total + 255) / 256, 256, 0, stream>>>(
      feats, fc2o, st_b3, g3, n3, 1.0f / M1, points1, (float*)d_out);
}
